// FlaxMixtralBlockSparesTop2MLPCollection_62062277427764
// MI455X (gfx1250) — matrix operand order repaired and checked by reading, not compiled, not run on hardware
//
#include <hip/hip_runtime.h>
#include <hip/hip_bf16.h>

// MoE (Mixtral top-2) fused MLP for MI455X / gfx1250.
// route/compact -> zero out -> GEMM1(silu*mul, bf16 WMMA) -> GEMM2(TDM + WMMA + scatter).
// Workspace (~59 MB): cnt/cur/off | tok[8192] | wt[8192] | g[8192*3584] bf16.

#define NUM_EXPERTS 8
#define TOPK 2
#define HIDDEN 1024
#define INTER 3584
#define TOKENS 4096
#define MT 128   // workgroup M tile (tokens)
#define NT 64    // workgroup N tile
#define KB 32    // K step per WMMA pass
#define KP 40    // padded LDS K stride (bf16 elems) = 80B, keeps 16B alignment

typedef __bf16 bf16;
typedef __attribute__((ext_vector_type(16))) __bf16 v16bf;
typedef __attribute__((ext_vector_type(8)))  __bf16 v8bf;
typedef __attribute__((ext_vector_type(2)))  __bf16 v2bf;
typedef __attribute__((ext_vector_type(8)))  float  v8f;
typedef __attribute__((ext_vector_type(4)))  float  v4f;
typedef __attribute__((ext_vector_type(4)))  unsigned v4u;
typedef __attribute__((ext_vector_type(2)))  unsigned v2u;
typedef __attribute__((ext_vector_type(4)))  unsigned u32x4;
typedef __attribute__((ext_vector_type(8)))  int      i32x8;
typedef __attribute__((ext_vector_type(4)))  int      i32x4;

__device__ __forceinline__ unsigned pk2(float lo, float hi) {
  v2bf p; p[0] = (__bf16)lo; p[1] = (__bf16)hi;   // v_cvt_pk_bf16_f32 path
  return __builtin_bit_cast(unsigned, p);
}

__device__ __forceinline__ v8f v8f_zero() {
  v8f z;
#pragma unroll
  for (int i = 0; i < 8; ++i) z[i] = 0.0f;
  return z;
}

// A fragment (16x32 bf16, MxK): lane l holds row l%16.
// lanes 0-15: K = 0..15; lanes 16-31: K = 16..31. Two 16B LDS loads.
__device__ __forceinline__ v16bf frag_a(const bf16* S, int row0, int lane) {
  const bf16* p = S + (row0 + (lane & 15)) * KP + ((lane & 16) ? 16 : 0);
  v8bf lo = *(const v8bf*)p;
  v8bf hi = *(const v8bf*)(p + 8);
  v16bf f;
#pragma unroll
  for (int i = 0; i < 8; ++i) { f[i] = lo[i]; f[i + 8] = hi[i]; }
  return f;
}

// B fragment (32x16 bf16, KxN): lane l holds col l%16; K contiguous in LDS.
__device__ __forceinline__ v16bf frag_b(const bf16* S, int col0, int lane) {
  const bf16* p = S + (col0 + (lane & 15)) * KP + ((lane & 16) ? 16 : 0);
  v8bf lo = *(const v8bf*)p;
  v8bf hi = *(const v8bf*)(p + 8);
  v16bf f;
#pragma unroll
  for (int i = 0; i < 8; ++i) { f[i] = lo[i]; f[i + 8] = hi[i]; }
  return f;
}

#define WMMA_BF16(A, B, C) \
  __builtin_amdgcn_wmma_f32_16x16x32_bf16(false, (A), false, (B), (short)0, (C), false, false)

// ---------------------------------------------------------------------------
// TDM: issue TENSOR_LOAD_TO_LDS of a 2D bf16 tile (rows x KB) from `gsrc`
// (row stride INTER elems) into LDS at `ldsaddr`, with LDS padding 16B per
// 64B row (reproduces the KP=40 layout). Rows >= rows_avail are zero-filled
// by TDM OOB handling. Wave-level op; call from one wave only.
// ---------------------------------------------------------------------------
__device__ __forceinline__ void tdm_load_tile(const bf16* gsrc, unsigned ldsaddr,
                                              unsigned rows_avail) {
  unsigned long long ga = (unsigned long long)(size_t)gsrc;
  u32x4 g0;
  g0[0] = 1u;                                   // count=1 valid descriptor
  g0[1] = ldsaddr;                              // lds_addr (bytes)
  g0[2] = (unsigned)ga;                         // global_addr[31:0]
  g0[3] = (unsigned)(ga >> 32) | (2u << 30);    // global_addr[56:32] | type=2
  i32x8 g1;
  g1[0] = (int)((1u << 16) |                    // data_size = 2B
                (1u << 20) |                    // pad_enable
                (3u << 22) |                    // pad_interval: 16 DWORDs (64B)
                (3u << 25));                    // pad_amount: 4 DWORDs (16B)
  g1[1] = (int)((INTER & 0xffffu) << 16);       // tensor_dim0[15:0]
  g1[2] = (int)((rows_avail & 0xffffu) << 16);  // dim0[31:16]=0 | tensor_dim1[15:0]
  g1[3] = (int)((rows_avail >> 16) | (KB << 16)); // tensor_dim1[31:16] | tile_dim0
  g1[4] = MT;                                   // tile_dim1 | tile_dim2=0
  g1[5] = INTER;                                // tensor_dim0_stride[31:0]
  g1[6] = 0;                                    // stride[47:32] | dim1_stride lo
  g1[7] = 0;
  i32x4 z; z[0] = 0; z[1] = 0; z[2] = 0; z[3] = 0;
#if __clang_major__ >= 23
  i32x8 z8;
#pragma unroll
  for (int i = 0; i < 8; ++i) z8[i] = 0;
  __builtin_amdgcn_tensor_load_to_lds(g0, g1, z, z, z8, 0);
#else
  __builtin_amdgcn_tensor_load_to_lds(g0, g1, z, z, 0);
#endif
}

// ---------------------------------------------------------------------------
// Kernel 1: routing compaction. One block.
// ---------------------------------------------------------------------------
__global__ void route_kernel(const int* __restrict__ mask, const float* __restrict__ rw,
                             int* __restrict__ cnt, int* __restrict__ cur,
                             int* __restrict__ off, int* __restrict__ tok,
                             float* __restrict__ wt) {
  const int tid = threadIdx.x;
  if (tid < NUM_EXPERTS) { cnt[tid] = 0; cur[tid] = 0; }
  __syncthreads();
  for (int idx = tid; idx < TOKENS * TOPK; idx += blockDim.x) {
    int t = idx >> 1, k = idx & 1;
#pragma unroll
    for (int e = 0; e < NUM_EXPERTS; ++e) {
      if (mask[(e * TOPK + k) * TOKENS + t]) { atomicAdd(&cnt[e], 1); break; }
    }
  }
  __threadfence();
  __syncthreads();
  if (tid == 0) {
    int s = 0;
    for (int e = 0; e < NUM_EXPERTS; ++e) { off[e] = s; s += cnt[e]; }
    off[NUM_EXPERTS] = s;
  }
  __threadfence();
  __syncthreads();
  for (int idx = tid; idx < TOKENS * TOPK; idx += blockDim.x) {
    int t = idx >> 1, k = idx & 1;
#pragma unroll
    for (int e = 0; e < NUM_EXPERTS; ++e) {
      if (mask[(e * TOPK + k) * TOKENS + t]) {
        int pos = atomicAdd(&cur[e], 1);
        tok[off[e] + pos] = t;
        wt[off[e] + pos]  = rw[t * TOPK + k];
        break;
      }
    }
  }
}

__global__ void zero_kernel(float4* __restrict__ o, int n4) {
  int i = blockIdx.x * blockDim.x + threadIdx.x;
  if (i < n4) o[i] = make_float4(0.f, 0.f, 0.f, 0.f);
}

// ---------------------------------------------------------------------------
// Kernel 2: g = silu(x@w1) * (x@w3).  Grid: (INTER/NT, TOKENS/MT, 8).
// Invalid rows load token 0 (garbage rows never stored: epilogue guards r<count).
// ---------------------------------------------------------------------------
__global__ __launch_bounds__(256, 2)
void moe_gemm1_silu(const float* __restrict__ hs, const float* __restrict__ w1,
                    const float* __restrict__ w3, const int* __restrict__ cnt,
                    const int* __restrict__ off, const int* __restrict__ tok,
                    bf16* __restrict__ g) {
  const int e  = blockIdx.z;
  const int m0 = blockIdx.y * MT;
  const int n0 = blockIdx.x * NT;
  const int count = cnt[e];
  if (m0 >= count) return;
  const int base = off[e];

  __shared__ __align__(16) bf16 As[MT * KP];
  __shared__ __align__(16) bf16 B1s[NT * KP];
  __shared__ __align__(16) bf16 B3s[NT * KP];
  __shared__ int stok[MT];

  const int tid  = threadIdx.x;
  const int lane = tid & 31;
  const int wid  = tid >> 5;
  const int wm   = (wid & 3) * 32;
  const int wn   = (wid >> 2) * 32;

  for (int i = tid; i < MT; i += 256) {
    int r = m0 + i;
    int t = (r < count) ? tok[base + r] : 0;   // clamp, no masking needed
    stok[i] = t;
  }

  v8f acc1[2][2], acc3[2][2];
#pragma unroll
  for (int i = 0; i < 2; ++i)
#pragma unroll
    for (int j = 0; j < 2; ++j) { acc1[i][j] = v8f_zero(); acc3[i][j] = v8f_zero(); }

  const float* w1e = w1 + (size_t)e * HIDDEN * INTER + n0;
  const float* w3e = w3 + (size_t)e * HIDDEN * INTER + n0;

  const int bk = (tid >> 4) * 2;   // 0..30, even
  const int bn = (tid & 15) * 4;   // 0..60
  v4f fa[4];
  v4f f1a, f1b, f3a, f3b;
  __syncthreads();

  auto fetch = [&](int k0) {
#pragma unroll
    for (int i = 0; i < 4; ++i) {
      int q = tid + 256 * i;
      int m = q >> 3, j = q & 7;
      fa[i] = *(const v4f*)(hs + (size_t)stok[m] * HIDDEN + k0 + 4 * j);
    }
    f1a = *(const v4f*)(w1e + (size_t)(k0 + bk)     * INTER + bn);
    f1b = *(const v4f*)(w1e + (size_t)(k0 + bk + 1) * INTER + bn);
    f3a = *(const v4f*)(w3e + (size_t)(k0 + bk)     * INTER + bn);
    f3b = *(const v4f*)(w3e + (size_t)(k0 + bk + 1) * INTER + bn);
  };
  auto stage = [&]() {
#pragma unroll
    for (int i = 0; i < 4; ++i) {
      int q = tid + 256 * i;
      int m = q >> 3, j = q & 7;
      v2u d;
      d.x = pk2(fa[i][0], fa[i][1]);
      d.y = pk2(fa[i][2], fa[i][3]);
      *(v2u*)&As[m * KP + 4 * j] = d;
    }
#pragma unroll
    for (int j = 0; j < 4; ++j) {
      *(unsigned*)&B1s[(bn + j) * KP + bk] = pk2(f1a[j], f1b[j]);
      *(unsigned*)&B3s[(bn + j) * KP + bk] = pk2(f3a[j], f3b[j]);
    }
  };

  fetch(0);
  for (int k0 = 0; k0 < HIDDEN; k0 += KB) {
    stage();
    __syncthreads();
    if (k0 + KB < HIDDEN) fetch(k0 + KB);   // next slice in flight during WMMA

    v16bf a0  = frag_a(As,  wm,      lane);
    v16bf a1  = frag_a(As,  wm + 16, lane);
    v16bf b10 = frag_b(B1s, wn,      lane);
    v16bf b11 = frag_b(B1s, wn + 16, lane);
    v16bf b30 = frag_b(B3s, wn,      lane);
    v16bf b31 = frag_b(B3s, wn + 16, lane);

    acc1[0][0] = WMMA_BF16(a0, b10, acc1[0][0]);
    acc1[0][1] = WMMA_BF16(a0, b11, acc1[0][1]);
    acc1[1][0] = WMMA_BF16(a1, b10, acc1[1][0]);
    acc1[1][1] = WMMA_BF16(a1, b11, acc1[1][1]);
    acc3[0][0] = WMMA_BF16(a0, b30, acc3[0][0]);
    acc3[0][1] = WMMA_BF16(a0, b31, acc3[0][1]);
    acc3[1][0] = WMMA_BF16(a1, b30, acc3[1][0]);
    acc3[1][1] = WMMA_BF16(a1, b31, acc3[1][1]);
    __syncthreads();
  }

  // fused silu * gate epilogue -> bf16 intermediate
#pragma unroll
  for (int mi = 0; mi < 2; ++mi)
#pragma unroll
    for (int ni = 0; ni < 2; ++ni) {
      int n  = n0 + wn + 16 * ni + (lane & 15);
      int rb = m0 + wm + 16 * mi + ((lane & 16) ? 8 : 0);
#pragma unroll
      for (int v = 0; v < 8; ++v) {
        int r = rb + v;
        if (r < count) {
          float x = acc1[mi][ni][v];
          float h = (x / (1.0f + __expf(-x))) * acc3[mi][ni][v];
          g[(size_t)(base + r) * INTER + n] = (__bf16)h;
        }
      }
    }
}

// ---------------------------------------------------------------------------
// Kernel 3: out[tok] += wt * (g @ w2).  Grid: (HIDDEN/NT, TOKENS/MT, 8).
// A tile streamed by the Tensor Data Mover into double-buffered LDS.
// ---------------------------------------------------------------------------
__global__ __launch_bounds__(256, 2)
void moe_gemm2_scatter(const bf16* __restrict__ g, const float* __restrict__ w2,
                       const int* __restrict__ cnt, const int* __restrict__ off,
                       const int* __restrict__ tok, const float* __restrict__ wt,
                       float* __restrict__ out) {
  const int e  = blockIdx.z;
  const int m0 = blockIdx.y * MT;
  const int n0 = blockIdx.x * NT;
  const int count = cnt[e];
  if (m0 >= count) return;
  const int base = off[e];

  __shared__ __align__(16) bf16 As[2 * MT * KP];   // TDM double buffer
  __shared__ __align__(16) bf16 Bs[NT * KP];
  __shared__ int   stok[MT];
  __shared__ float swt[MT];

  const int tid  = threadIdx.x;
  const int lane = tid & 31;
  const int wid  = tid >> 5;
  const int wm   = (wid & 3) * 32;
  const int wn   = (wid >> 2) * 32;

  for (int i = tid; i < MT; i += 256) {
    int r = m0 + i;
    stok[i] = (r < count) ? tok[base + r] : 0;
    swt[i]  = (r < count) ? wt[base + r] : 0.0f;
  }

  v8f acc[2][2];
#pragma unroll
  for (int i = 0; i < 2; ++i)
#pragma unroll
    for (int j = 0; j < 2; ++j) acc[i][j] = v8f_zero();

  const float* w2e = w2 + (size_t)e * INTER * HIDDEN + n0;
  const bf16* gA   = g + (size_t)(base + m0) * INTER;   // tile origin
  const unsigned rows_avail = (unsigned)(count - m0);   // TDM zero-fills beyond
  const unsigned lds0 = (unsigned)(size_t)&As[0];
  const unsigned lds1 = (unsigned)(size_t)&As[MT * KP];

  const int bk = (tid >> 4) * 2;
  const int bn = (tid & 15) * 4;
  v4f fba, fbb;
  __syncthreads();

  auto fetchB = [&](int k0) {
    fba = *(const v4f*)(w2e + (size_t)(k0 + bk)     * HIDDEN + bn);
    fbb = *(const v4f*)(w2e + (size_t)(k0 + bk + 1) * HIDDEN + bn);
  };
  auto stageB = [&]() {
#pragma unroll
    for (int j = 0; j < 4; ++j)
      *(unsigned*)&Bs[(bn + j) * KP + bk] = pk2(fba[j], fbb[j]);
  };

  if (wid == 0) tdm_load_tile(gA, lds0, rows_avail);    // TENSOR_LOAD_TO_LDS
  fetchB(0);
  unsigned p = 0;
  for (int k0 = 0; k0 < INTER; k0 += KB) {
    stageB();
    if (wid == 0) __builtin_amdgcn_s_wait_tensorcnt(0); // A(p) tile landed
    __syncthreads();
    if (k0 + KB < INTER) {
      if (wid == 0)
        tdm_load_tile(gA + (k0 + KB), p ? lds0 : lds1, rows_avail);
      fetchB(k0 + KB);
    }

    const bf16* Ap = &As[p * MT * KP];
    v16bf a0 = frag_a(Ap, wm,      lane);
    v16bf a1 = frag_a(Ap, wm + 16, lane);
    v16bf b0 = frag_b(Bs, wn,      lane);
    v16bf b1 = frag_b(Bs, wn + 16, lane);

    acc[0][0] = WMMA_BF16(a0, b0, acc[0][0]);
    acc[0][1] = WMMA_BF16(a0, b1, acc[0][1]);
    acc[1][0] = WMMA_BF16(a1, b0, acc[1][0]);
    acc[1][1] = WMMA_BF16(a1, b1, acc[1][1]);
    __syncthreads();
    p ^= 1;
  }

  // scale by routing weight, scatter-add (each token hits 2 experts)
#pragma unroll
  for (int mi = 0; mi < 2; ++mi)
#pragma unroll
    for (int ni = 0; ni < 2; ++ni) {
      int n   = n0 + wn + 16 * ni + (lane & 15);
      int rtb = wm + 16 * mi + ((lane & 16) ? 8 : 0);
#pragma unroll
      for (int v = 0; v < 8; ++v) {
        int rt = rtb + v;
        if (m0 + rt < count) {
          float val = acc[mi][ni][v] * swt[rt];
          atomicAdd(&out[(size_t)stok[rt] * HIDDEN + n], val);
        }
      }
    }
}

// ---------------------------------------------------------------------------
extern "C" void kernel_launch(void* const* d_in, const int* in_sizes, int n_in,
                              void* d_out, int out_size, void* d_ws, size_t ws_size,
                              hipStream_t stream) {
  const int*   mask = (const int*)d_in[0];    // (8, 2, 4096)
  const float* hs   = (const float*)d_in[1];  // (4096, 1024)
  const float* rw   = (const float*)d_in[2];  // (4096, 2)
  const float* w1   = (const float*)d_in[3];  // (8, 1024, 3584)
  const float* w2   = (const float*)d_in[4];  // (8, 3584, 1024)
  const float* w3   = (const float*)d_in[5];  // (8, 1024, 3584)
  float* out = (float*)d_out;                 // (4096, 1024)

  char* p   = (char*)d_ws;
  int*  cnt = (int*)p;            // 8
  int*  cur = cnt + 8;            // 8
  int*  off = cur + 8;            // 9
  int*  tok = (int*)(p + 1024);                                    // 8192 ints
  float* wt = (float*)(p + 1024 + TOKENS * TOPK * sizeof(int));    // 8192 floats
  bf16* g   = (bf16*)(p + 1024 + 2 * TOKENS * TOPK * sizeof(int)); // 8192*3584 bf16
  (void)in_sizes; (void)n_in; (void)out_size; (void)ws_size;

  route_kernel<<<1, 256, 0, stream>>>(mask, rw, cnt, cur, off, tok, wt);

  zero_kernel<<<(TOKENS * HIDDEN / 4 + 255) / 256, 256, 0, stream>>>(
      (float4*)out, TOKENS * HIDDEN / 4);

  dim3 g1(INTER / NT, TOKENS / MT, NUM_EXPERTS);
  moe_gemm1_silu<<<g1, 256, 0, stream>>>(hs, w1, w3, cnt, off, tok, g);

  dim3 g2(HIDDEN / NT, TOKENS / MT, NUM_EXPERTS);
  moe_gemm2_scatter<<<g2, 256, 0, stream>>>(g, w2, cnt, off, tok, wt, out);
}
